// LLA_27702539059803
// MI455X (gfx1250) — matrix operand order repaired and checked by reading, not compiled, not run on hardware
//
#include <hip/hip_runtime.h>
#include <math.h>
#include <stdint.h>

typedef __bf16 bf16;
typedef __attribute__((ext_vector_type(16))) __bf16 v16bf;
typedef __attribute__((ext_vector_type(8)))  __bf16 v8bf;
typedef __attribute__((ext_vector_type(8)))  float  v8f;

#define HH 56
#define WW 56
#define CC 128
#define NPIX (HH*WW)     // 3136
#define EPSBN 1e-5f

// ---------------------------------------------------------------------------
// WMMA operand loaders (wave32, CDNA5 16x16x32 bf16 layouts, CDNA5 ISA)
// A 16x32 (MxK): lane l (<16): M=l, e0..7 -> K 0..7, e8..15 -> K 16..23
//                lane l (>=16): M=l-16, e0..7 -> K 8..15, e8..15 -> K 24..31
// B 32x16 (KxN) stored [N][K]: lane l (<16): N=l, e0..7 -> K 0..7, e8..15 -> K 16..23 ; lane>=16: N=l-16, e0..7 -> K 8..15, e8..15 -> K 24..31
// C/D 16x16 f32: elem r: lane<16 -> (M=r, N=lane); lane>=16 -> (M=r+8, N=lane-16)
// ---------------------------------------------------------------------------
__device__ __forceinline__ v16bf mk16(v8bf lo, v8bf hi) {
  v16bf r;
#pragma unroll
  for (int i = 0; i < 8; ++i) { r[i] = lo[i]; r[i + 8] = hi[i]; }
  return r;
}

// rowp already points at row (tile_m*16 + (lane&15)) of an [*, strideK] bf16 array
__device__ __forceinline__ v16bf ldA(const bf16* rowp, int lane, int ktile) {
  int off = ktile * 32 + ((lane < 16) ? 0 : 8);
  v8bf lo = *(const v8bf*)(rowp + off);
  v8bf hi = *(const v8bf*)(rowp + off + 16);
  return mk16(lo, hi);
}

// wt is [N][K] bf16, row stride = K
__device__ __forceinline__ v16bf ldB(const bf16* wt, int lane, int ntile, int ktile, int K) {
  const bf16* p = wt + (size_t)(ntile * 16 + (lane & 15)) * K + ktile * 32 + ((lane < 16) ? 0 : 8);
  v8bf lo = *(const v8bf*)(p);
  v8bf hi = *(const v8bf*)(p + 16);
  return mk16(lo, hi);
}

__device__ __forceinline__ float gelu_f(float v) {
  return 0.5f * v * (1.0f + erff(v * 0.70710678118654752f));
}

// ---------------------------------------------------------------------------
// prep: xbf = bf16(x); h1 = roll(+c along H); g1 = roll(-c along W)
// ---------------------------------------------------------------------------
__global__ void prep_kernel(const float* __restrict__ x, bf16* __restrict__ xbf,
                            bf16* __restrict__ h1, bf16* __restrict__ g1, int total) {
  int idx = blockIdx.x * blockDim.x + threadIdx.x;
  if (idx >= total) return;
  int c = idx & 127;
  int m = idx >> 7;
  int b = m / NPIX;
  int n = m - b * NPIX;
  int h = n / WW;
  int w = n - h * WW;
  xbf[idx] = (bf16)x[idx];
  int hs = (h - c) % HH; if (hs < 0) hs += HH;          // out[h] = in[(h-c)%H]
  h1[idx] = (bf16)x[((size_t)(b * NPIX + hs * WW + w)) * CC + c];
  int wsrc = (w + c) % WW;                               // shift -c: out[w] = in[(w+c)%W]
  g1[idx] = (bf16)x[((size_t)(b * NPIX + h * WW + wsrc)) * CC + c];
}

// h2 = roll t1 by +c along W ; g2 = roll t2 by +c along H
__global__ void roll2_kernel(const bf16* __restrict__ t1, const bf16* __restrict__ t2,
                             bf16* __restrict__ h2, bf16* __restrict__ g2, int total) {
  int idx = blockIdx.x * blockDim.x + threadIdx.x;
  if (idx >= total) return;
  int c = idx & 127;
  int m = idx >> 7;
  int b = m / NPIX;
  int n = m - b * NPIX;
  int h = n / WW;
  int w = n - h * WW;
  int wsrc = (w - c) % WW; if (wsrc < 0) wsrc += WW;
  h2[idx] = t1[((size_t)(b * NPIX + h * WW + wsrc)) * CC + c];
  int hs = (h - c) % HH; if (hs < 0) hs += HH;
  g2[idx] = t2[((size_t)(b * NPIX + hs * WW + w)) * CC + c];
}

// ---------------------------------------------------------------------------
// weight converters: transpose [K,N] f32 -> [N][K] bf16 ; plain convert
// ---------------------------------------------------------------------------
__global__ void wcvt_t(const float* __restrict__ src, bf16* __restrict__ dst, int K, int N) {
  int idx = blockIdx.x * blockDim.x + threadIdx.x;
  if (idx >= K * N) return;
  int n = idx % N;
  int k = idx / N;
  dst[(size_t)n * K + k] = (bf16)src[(size_t)k * N + n];
}
__global__ void wcvt_c(const float* __restrict__ src, bf16* __restrict__ dst, int total) {
  int idx = blockIdx.x * blockDim.x + threadIdx.x;
  if (idx < total) dst[idx] = (bf16)src[idx];
}

// ---------------------------------------------------------------------------
// Tiled WMMA GEMM: out[M,128] = epi(A[M,K] @ Wt^T + bias [+ resid])
// Wt stored [128][K] bf16, staged into LDS once per block via
// global_load_async_to_lds_b128 (ASYNCcnt path), then consumed by WMMA via
// ds_load_b128. Block = 8 waves, each wave -> 16 rows x 128 cols.
// EPI: 0 = GELU -> bf16 ; 1 = bias -> bf16 ; 2 = +resid -> f32 ; 3 = bias -> f32
// ---------------------------------------------------------------------------
template <int KT, int EPI>
__global__ __launch_bounds__(256) void gemm_kernel(const bf16* __restrict__ A,
                                                   const bf16* __restrict__ Wt,
                                                   const float* __restrict__ bias,
                                                   const float* __restrict__ resid,
                                                   void* __restrict__ outp, int M) {
  const int K = KT * 32;
  extern __shared__ __align__(16) char smem[];         // K*256 bytes (Wt tile)
  bf16* lWt = (bf16*)smem;

  int lane = threadIdx.x;
  int wave = threadIdx.y;
  int tid = wave * 32 + lane;

  // --- async stage of the full [128][K] weight tile into LDS (16B granules)
  {
    const int chunks = (K * 256) / 16;                 // 2048 (K=128) / 4096 (K=256)
    uint32_t lds_base = (uint32_t)(uintptr_t)smem;     // LDS aperture: addr[31:0]
    uint64_t gbase = (uint64_t)(uintptr_t)Wt;
#pragma unroll
    for (int i = 0; i < chunks / 256; ++i) {
      uint32_t boff = (uint32_t)(tid + i * 256) * 16u;
      asm volatile("global_load_async_to_lds_b128 %0, %1, %2 offset:0"
                   :
                   : "v"(lds_base + boff), "v"(boff), "s"(gbase)
                   : "memory");
    }
    asm volatile("s_wait_asynccnt 0" ::: "memory");
  }
  __syncthreads();

  int m0 = (blockIdx.x * 8 + wave) * 16;
  if (m0 < M) {
    const bf16* rowp = A + (size_t)(m0 + (lane & 15)) * K;
    if (m0 + 128 < M) __builtin_prefetch(rowp + (size_t)128 * K, 0, 1);  // next block's A

    v16bf a[KT];
#pragma unroll
    for (int k = 0; k < KT; ++k) a[k] = ldA(rowp, lane, k);

    int col_lane = lane & 15;
    int moff = (lane < 16) ? 0 : 8;

#pragma unroll
    for (int nt = 0; nt < 8; ++nt) {
      v8f acc = {};
#pragma unroll
      for (int k = 0; k < KT; ++k) {
        v16bf b = ldB(lWt, lane, nt, k, K);
        acc = __builtin_amdgcn_wmma_f32_16x16x32_bf16(false, a[k], false, b,
                                                      (short)0, acc, false, false);
      }
      int col = nt * 16 + col_lane;
      float bv = bias[col];
#pragma unroll
      for (int r = 0; r < 8; ++r) {
        size_t m = (size_t)(m0 + moff + r);
        float v = acc[r] + bv;
        if (EPI == 0) {
          ((bf16*)outp)[m * 128 + col] = (bf16)gelu_f(v);
        } else if (EPI == 1) {
          ((bf16*)outp)[m * 128 + col] = (bf16)v;
        } else if (EPI == 2) {
          ((float*)outp)[m * 128 + col] = v + resid[m * 128 + col];
        } else {
          ((float*)outp)[m * 128 + col] = v;
        }
      }
    }
  }
}

// ---------------------------------------------------------------------------
// Window attention: 1 block (4 waves) per 7x7 window, padded to 64 tokens.
// S = Q K^T (WMMA), masked softmax (lane shuffles), O = P V (WMMA) -> f32 att
// ---------------------------------------------------------------------------
__global__ __launch_bounds__(128) void attn_kernel(const bf16* __restrict__ q,
                                                   const bf16* __restrict__ k,
                                                   const bf16* __restrict__ v,
                                                   float* __restrict__ att) {
  __shared__ __align__(16) bf16 Qs[64 * 128];
  __shared__ __align__(16) bf16 Ks[64 * 128];
  __shared__ __align__(16) bf16 Vt[128 * 64];  // [ch][token]
  __shared__ __align__(16) bf16 Ps[64 * 64];

  int tid = threadIdx.x;
  int win = blockIdx.x;
  int b = win >> 6;
  int wy = (win >> 3) & 7;
  int wx = win & 7;

  {  // stage Q,K,V^T into LDS; tokens >= 49 zero-padded
    int t = tid >> 1, half = tid & 1;
    int ch0 = half * 64;
    bool valid = (t < 49);
    size_t grow = 0;
    if (valid) {
      int h = wy * 7 + t / 7, wc = wx * 7 + t % 7;
      grow = ((size_t)b * NPIX + h * WW + wc) * CC;
    }
    v8bf z = {};
#pragma unroll
    for (int i = 0; i < 8; ++i) {
      v8bf qv = valid ? *(const v8bf*)(q + grow + ch0 + i * 8) : z;
      v8bf kv = valid ? *(const v8bf*)(k + grow + ch0 + i * 8) : z;
      v8bf vv = valid ? *(const v8bf*)(v + grow + ch0 + i * 8) : z;
      *(v8bf*)(Qs + t * 128 + ch0 + i * 8) = qv;
      *(v8bf*)(Ks + t * 128 + ch0 + i * 8) = kv;
#pragma unroll
      for (int j = 0; j < 8; ++j) Vt[(ch0 + i * 8 + j) * 64 + t] = vv[j];
    }
  }
  __syncthreads();

  int lane = tid & 31;
  int mtile = tid >> 5;  // wave id: query rows [mtile*16, mtile*16+16)
  int col_lane = lane & 15;
  int moff = (lane < 16) ? 0 : 8;
  const float scale = 0.08838834764831845f;  // 128^-0.5

  v16bf aQ[4];
  const bf16* qrow = Qs + (size_t)(mtile * 16 + col_lane) * 128;
#pragma unroll
  for (int kt = 0; kt < 4; ++kt) aQ[kt] = ldA(qrow, lane, kt);

  v8f S[4];
#pragma unroll
  for (int nt = 0; nt < 4; ++nt) {
    v8f acc = {};
#pragma unroll
    for (int kt = 0; kt < 4; ++kt) {
      v16bf bK = ldB(Ks, lane, nt, kt, 128);
      acc = __builtin_amdgcn_wmma_f32_16x16x32_bf16(false, aQ[kt], false, bK,
                                                    (short)0, acc, false, false);
    }
    S[nt] = acc;
  }

  bool cval[4];
#pragma unroll
  for (int nt = 0; nt < 4; ++nt) cval[nt] = (nt * 16 + col_lane) < 49;

#pragma unroll
  for (int r = 0; r < 8; ++r) {
    float sv[4], pv[4];
    float mx = -1e30f;
#pragma unroll
    for (int nt = 0; nt < 4; ++nt) {
      sv[nt] = cval[nt] ? S[nt][r] * scale : -1e30f;
      mx = fmaxf(mx, sv[nt]);
    }
    mx = fmaxf(mx, __shfl_xor(mx, 1, 32));
    mx = fmaxf(mx, __shfl_xor(mx, 2, 32));
    mx = fmaxf(mx, __shfl_xor(mx, 4, 32));
    mx = fmaxf(mx, __shfl_xor(mx, 8, 32));
    float sum = 0.f;
#pragma unroll
    for (int nt = 0; nt < 4; ++nt) {
      pv[nt] = cval[nt] ? __expf(sv[nt] - mx) : 0.f;
      sum += pv[nt];
    }
    sum += __shfl_xor(sum, 1, 32);
    sum += __shfl_xor(sum, 2, 32);
    sum += __shfl_xor(sum, 4, 32);
    sum += __shfl_xor(sum, 8, 32);
    float inv = 1.f / sum;
    int prow = mtile * 16 + moff + r;
#pragma unroll
    for (int nt = 0; nt < 4; ++nt)
      Ps[prow * 64 + nt * 16 + col_lane] = (bf16)(pv[nt] * inv);
  }
  __syncthreads();

  v16bf aP[2];
  const bf16* prow = Ps + (size_t)(mtile * 16 + col_lane) * 64;
#pragma unroll
  for (int kt = 0; kt < 2; ++kt) aP[kt] = ldA(prow, lane, kt);

#pragma unroll
  for (int nt = 0; nt < 8; ++nt) {
    v8f acc = {};
#pragma unroll
    for (int kt = 0; kt < 2; ++kt) {
      v16bf bV = ldB(Vt, lane, nt, kt, 64);
      acc = __builtin_amdgcn_wmma_f32_16x16x32_bf16(false, aP[kt], false, bV,
                                                    (short)0, acc, false, false);
    }
    int ch = nt * 16 + col_lane;
#pragma unroll
    for (int r = 0; r < 8; ++r) {
      int tok = mtile * 16 + moff + r;
      if (tok < 49) {
        int h = wy * 7 + tok / 7, wc = wx * 7 + tok % 7;
        att[((size_t)b * NPIX + h * WW + wc) * CC + ch] = acc[r];
      }
    }
  }
}

// ---------------------------------------------------------------------------
// LayerNorm over concat(x1,x2) [2C=256] -> bf16 cat ; 1 wave per row
// ---------------------------------------------------------------------------
__global__ void ln_kernel(const float* __restrict__ x1, const float* __restrict__ x2,
                          const float* __restrict__ lnw, const float* __restrict__ lnb,
                          bf16* __restrict__ cat, int M) {
  int lane = threadIdx.x & 31;
  int wave = threadIdx.x >> 5;
  int row = blockIdx.x * 8 + wave;
  if (row >= M) return;
  const float* src = (lane < 16) ? (x1 + (size_t)row * 128 + lane * 8)
                                 : (x2 + (size_t)row * 128 + (lane - 16) * 8);
  float vals[8];
  float s = 0.f, sq = 0.f;
#pragma unroll
  for (int i = 0; i < 8; ++i) { vals[i] = src[i]; s += vals[i]; sq += vals[i] * vals[i]; }
#pragma unroll
  for (int m = 1; m < 32; m <<= 1) { s += __shfl_xor(s, m, 32); sq += __shfl_xor(sq, m, 32); }
  float mean = s * (1.f / 256.f);
  float var = sq * (1.f / 256.f) - mean * mean;
  float inv = rsqrtf(var + EPSBN);
  int e0 = lane * 8;
#pragma unroll
  for (int i = 0; i < 8; ++i) {
    int e = e0 + i;
    cat[(size_t)row * 256 + e] = (bf16)((vals[i] - mean) * inv * lnw[e] + lnb[e]);
  }
}

// ---------------------------------------------------------------------------
// BatchNorm helpers (deterministic two-stage column reduction, no atomics)
// ---------------------------------------------------------------------------
__global__ void colsum_kernel(const float* __restrict__ src, float* __restrict__ psum,
                              float* __restrict__ psq, int M) {
  int c = threadIdx.x;  // 128
  int rows = M / gridDim.x;
  int r0 = blockIdx.x * rows;
  float s = 0.f, q = 0.f;
  for (int r = r0; r < r0 + rows; ++r) {
    float v = src[(size_t)r * 128 + c];
    s += v;
    q += v * v;
  }
  psum[blockIdx.x * 128 + c] = s;
  psq[blockIdx.x * 128 + c] = q;
}

__global__ void reduce_stats(const float* __restrict__ psum, const float* __restrict__ psq,
                             float* __restrict__ s, float* __restrict__ q, int nblk) {
  int c = threadIdx.x;  // 128
  float a = 0.f, b = 0.f;
  for (int i = 0; i < nblk; ++i) { a += psum[i * 128 + c]; b += psq[i * 128 + c]; }
  s[c] = a;
  q[c] = b;
}

__global__ void bnrelu_kernel(const float* __restrict__ att, const float* __restrict__ s1,
                              const float* __restrict__ q1, const float* __restrict__ g,
                              const float* __restrict__ beta, float* __restrict__ y,
                              int M, int total) {
  int idx = blockIdx.x * blockDim.x + threadIdx.x;
  if (idx >= total) return;
  int c = idx & 127;
  float invM = 1.f / (float)M;
  float mu = s1[c] * invM;
  float var = q1[c] * invM - mu * mu;
  float v = (att[idx] - mu) * rsqrtf(var + EPSBN) * g[c] + beta[c];
  y[idx] = fmaxf(v, 0.f);
}

__global__ void cat2_kernel(const float* __restrict__ x1, const float* __restrict__ y,
                            const float* __restrict__ s2, const float* __restrict__ q2,
                            const float* __restrict__ g, const float* __restrict__ beta,
                            bf16* __restrict__ cat, int M, int total) {
  int idx = blockIdx.x * blockDim.x + threadIdx.x;
  if (idx >= total) return;
  int c = idx & 127;
  int m = idx >> 7;
  float invM = 1.f / (float)M;
  float mu = s2[c] * invM;
  float var = q2[c] * invM - mu * mu;
  float v = (y[idx] - mu) * rsqrtf(var + EPSBN) * g[c] + beta[c];
  cat[(size_t)m * 256 + c] = (bf16)x1[idx];
  cat[(size_t)m * 256 + 128 + c] = (bf16)v;
}

// ---------------------------------------------------------------------------
extern "C" void kernel_launch(void* const* d_in, const int* in_sizes, int n_in,
                              void* d_out, int out_size, void* d_ws, size_t ws_size,
                              hipStream_t stream) {
  const float* x     = (const float*)d_in[0];
  const float* fc1_w = (const float*)d_in[1];  const float* fc1_b = (const float*)d_in[2];
  const float* fc2_w = (const float*)d_in[3];  const float* fc2_b = (const float*)d_in[4];
  const float* fc3_w = (const float*)d_in[5];  const float* fc3_b = (const float*)d_in[6];
  const float* fc4_w = (const float*)d_in[7];  const float* fc4_b = (const float*)d_in[8];
  const float* fc5_w = (const float*)d_in[9];  const float* fc5_b = (const float*)d_in[10];
  const float* fc6_w = (const float*)d_in[11]; const float* fc6_b = (const float*)d_in[12];
  const float* ln_w  = (const float*)d_in[13]; const float* ln_b  = (const float*)d_in[14];
  const float* q_w   = (const float*)d_in[15]; const float* q_b   = (const float*)d_in[16];
  const float* k_w   = (const float*)d_in[17]; const float* k_b   = (const float*)d_in[18];
  const float* v_w   = (const float*)d_in[19]; const float* v_b   = (const float*)d_in[20];
  const float* bn1_g = (const float*)d_in[21]; const float* bn1_b = (const float*)d_in[22];
  const float* bn2_g = (const float*)d_in[23]; const float* bn2_b = (const float*)d_in[24];

  const int M = in_sizes[0] / CC;        // B * 3136 = 100352
  const int Bb = M / NPIX;               // 32
  const int total = M * CC;              // 12.8M elements
  const int nwin = Bb * 64;              // 2048 windows

  char* base = (char*)d_ws;
  size_t off = 0;
  auto alloc = [&](size_t bytes) -> void* {
    off = (off + 255) & ~(size_t)255;
    void* p = base + off;
    off += bytes;
    return p;
  };

  const size_t szb = (size_t)M * 128 * sizeof(bf16);
  const size_t szf = (size_t)M * 128 * sizeof(float);

  bf16* xbf  = (bf16*)alloc(szb);
  bf16* h1b  = (bf16*)alloc(szb);        // h1 -> h2 -> v-proj
  bf16* g1b  = (bf16*)alloc(szb);        // g1 -> g2
  bf16* t1b  = (bf16*)alloc(szb);        // t1 -> q-proj
  bf16* t2b  = (bf16*)alloc(szb);        // t2 -> k-proj
  bf16* catb = (bf16*)alloc((size_t)M * 256 * sizeof(bf16));
  float* fA  = (float*)alloc(szf);       // x_1 -> x1 (fc5 out)
  float* fB  = (float*)alloc(szf);       // x_2 -> y (relu(bn1))
  float* fC  = (float*)alloc(szf);       // att
  bf16* wfc1 = (bf16*)alloc(128 * 128 * sizeof(bf16));
  bf16* wfc2 = (bf16*)alloc(128 * 128 * sizeof(bf16));
  bf16* wfc3 = (bf16*)alloc(128 * 128 * sizeof(bf16));
  bf16* wfc4 = (bf16*)alloc(128 * 128 * sizeof(bf16));
  bf16* wfc5 = (bf16*)alloc(128 * 256 * sizeof(bf16));
  bf16* wfc6 = (bf16*)alloc(128 * 256 * sizeof(bf16));
  bf16* wq   = (bf16*)alloc(128 * 128 * sizeof(bf16));
  bf16* wk   = (bf16*)alloc(128 * 128 * sizeof(bf16));
  bf16* wv   = (bf16*)alloc(128 * 128 * sizeof(bf16));
  float* ps1 = (float*)alloc(128 * 128 * sizeof(float));
  float* pq1 = (float*)alloc(128 * 128 * sizeof(float));
  float* ps2 = (float*)alloc(128 * 128 * sizeof(float));
  float* pq2 = (float*)alloc(128 * 128 * sizeof(float));
  float* s1  = (float*)alloc(128 * sizeof(float));
  float* q1  = (float*)alloc(128 * sizeof(float));
  float* s2  = (float*)alloc(128 * sizeof(float));
  float* q2  = (float*)alloc(128 * sizeof(float));
  (void)ws_size; (void)n_in; (void)out_size;

  const int EB = 256;
  dim3 gblk((unsigned)(M / 128));
  dim3 gthr(32, 8);
  const size_t lds128 = 128 * 128 * sizeof(bf16);   // 32 KB weight tile
  const size_t lds256 = 128 * 256 * sizeof(bf16);   // 64 KB weight tile

  // 1. rolls + bf16 copy
  prep_kernel<<<(total + EB - 1) / EB, EB, 0, stream>>>(x, xbf, h1b, g1b, total);

  // 2. weights -> bf16 [N][K]
  wcvt_t<<<64, EB, 0, stream>>>(fc1_w, wfc1, 128, 128);
  wcvt_t<<<64, EB, 0, stream>>>(fc2_w, wfc2, 128, 128);
  wcvt_t<<<64, EB, 0, stream>>>(fc3_w, wfc3, 128, 128);
  wcvt_t<<<64, EB, 0, stream>>>(fc4_w, wfc4, 128, 128);
  wcvt_t<<<128, EB, 0, stream>>>(fc5_w, wfc5, 256, 128);
  wcvt_t<<<128, EB, 0, stream>>>(fc6_w, wfc6, 256, 128);
  wcvt_c<<<64, EB, 0, stream>>>(q_w, wq, 128 * 128);   // q/k/v: [out][in] == [N][K]
  wcvt_c<<<64, EB, 0, stream>>>(k_w, wk, 128 * 128);
  wcvt_c<<<64, EB, 0, stream>>>(v_w, wv, 128 * 128);

  // 3. MLP-roll branches
  gemm_kernel<4, 0><<<gblk, gthr, lds128, stream>>>(h1b, wfc1, fc1_b, nullptr, t1b, M);  // GELU
  gemm_kernel<4, 0><<<gblk, gthr, lds128, stream>>>(g1b, wfc3, fc3_b, nullptr, t2b, M);  // GELU
  roll2_kernel<<<(total + EB - 1) / EB, EB, 0, stream>>>(t1b, t2b, h1b, g1b, total);
  gemm_kernel<4, 2><<<gblk, gthr, lds128, stream>>>(h1b, wfc2, fc2_b, x, fA, M);  // x_1 = .. + x
  gemm_kernel<4, 2><<<gblk, gthr, lds128, stream>>>(g1b, wfc4, fc4_b, x, fB, M);  // x_2 = .. + x

  // 4. q/k/v projections + window attention
  gemm_kernel<4, 1><<<gblk, gthr, lds128, stream>>>(xbf, wq, q_b, nullptr, t1b, M);
  gemm_kernel<4, 1><<<gblk, gthr, lds128, stream>>>(xbf, wk, k_b, nullptr, t2b, M);
  gemm_kernel<4, 1><<<gblk, gthr, lds128, stream>>>(xbf, wv, v_b, nullptr, h1b, M);
  attn_kernel<<<nwin, 128, 0, stream>>>(t1b, t2b, h1b, fC);

  // 5. LN(cat(x_1,x_2)) -> fc5 -> + x
  ln_kernel<<<M / 8, EB, 0, stream>>>(fA, fB, ln_w, ln_b, catb, M);
  gemm_kernel<8, 2><<<gblk, gthr, lds256, stream>>>(catb, wfc5, fc5_b, x, fA, M);

  // 6. BN1 -> relu -> BN2
  colsum_kernel<<<128, 128, 0, stream>>>(fC, ps1, pq1, M);
  reduce_stats<<<1, 128, 0, stream>>>(ps1, pq1, s1, q1, 128);
  bnrelu_kernel<<<(total + EB - 1) / EB, EB, 0, stream>>>(fC, s1, q1, bn1_g, bn1_b, fB, M, total);
  colsum_kernel<<<128, 128, 0, stream>>>(fB, ps2, pq2, M);
  reduce_stats<<<1, 128, 0, stream>>>(ps2, pq2, s2, q2, 128);
  cat2_kernel<<<(total + EB - 1) / EB, EB, 0, stream>>>(fA, fB, s2, q2, bn2_g, bn2_b, catb, M, total);

  // 7. final projection
  gemm_kernel<8, 3><<<gblk, gthr, lds256, stream>>>(catb, wfc6, fc6_b, nullptr, (float*)d_out, M);
}
